// HierarchicalRouter_24215025615340
// MI455X (gfx1250) — matrix operand order repaired and checked by reading, not compiled, not run on hardware
//
#include <hip/hip_runtime.h>
#include <hip/hip_bf16.h>
#include <cstdint>
#include <cstddef>

typedef __attribute__((ext_vector_type(16))) _Float16 v16h;
typedef __attribute__((ext_vector_type(8)))  _Float16 v8h;
typedef __attribute__((ext_vector_type(4)))  _Float16 v4h;
typedef __attribute__((ext_vector_type(8)))  float    v8f;

#define H_      1024
#define S_      2048
#define B_      2
#define NT_     4096        // B*S tokens
#define NHEADS_ 8
#define DHEAD_  128
#define NE_     64
#define NH_     2
#define EPH_    32

// ---------------------------------------------------------------- WMMA core
__device__ __forceinline__ v8f wmma32(v16h a, v16h b, v8f c) {
    // D = A(16x32 f16) * B(32x16 f16) + C(16x16 f32)
    return __builtin_amdgcn_wmma_f32_16x16x32_f16(false, a, false, b,
                                                  (short)0, c, false, false);
}

// A fragment: 16x32 (MxK) from row-major [*,K] matrix, ld elements.
// Lane(r=lane&15, hi=lane>>4): row r, K = hi*8+{0..7} then 16+hi*8+{0..7}.
__device__ __forceinline__ v16h load_a(const _Float16* p, int row0, int k0, int ld) {
    int lane = threadIdx.x & 31;
    int r = lane & 15, hi = lane >> 4;
    const _Float16* base = p + (size_t)(row0 + r) * ld + k0 + hi * 8;
    union { v16h v; v8h h[2]; } u;
    u.h[0] = *(const v8h*)(base);
    u.h[1] = *(const v8h*)(base + 16);
    return u.v;
}

// B fragment: 32x16 (KxN) sourced from row-major [N,K] matrix (i.e. B^T rows).
// Lane(r,hi): column r, K = hi*8+{0..7} then 16+hi*8+{0..7}.
__device__ __forceinline__ v16h load_b(const _Float16* p, int row0, int k0, int ld) {
    int lane = threadIdx.x & 31;
    int r = lane & 15, hi = lane >> 4;
    const _Float16* base = p + (size_t)(row0 + r) * ld + k0 + hi * 8;
    union { v16h v; v8h h[2]; } u;
    u.h[0] = *(const v8h*)(base);
    u.h[1] = *(const v8h*)(base + 16);
    return u.v;
}

// ---------------------------------------------------------------- reductions
__device__ __forceinline__ float warp_sum32(float v) {
#pragma unroll
    for (int off = 16; off; off >>= 1) v += __shfl_xor(v, off, 32);
    return v;
}

__device__ __forceinline__ float block_sum8(float v, float* red) {
    int lane = threadIdx.x & 31, wid = threadIdx.x >> 5;
    v = warp_sum32(v);
    __syncthreads();
    if (lane == 0) red[wid] = v;
    __syncthreads();
    float s = (lane < 8) ? red[lane] : 0.0f;
#pragma unroll
    for (int off = 4; off; off >>= 1) s += __shfl_xor(s, off, 32);
    return __shfl(s, 0, 32);
}

// ---------------------------------------------------------------- kernels
__global__ void cvt_f32_to_f16(const float* __restrict__ in,
                               _Float16* __restrict__ outp, int n) {
    int i = blockIdx.x * blockDim.x + threadIdx.x;
    if (i < n) outp[i] = (_Float16)in[i];
}

// LayerNorm + top-router softmax, one block per token.
__global__ __launch_bounds__(256) void ln_top_kernel(
        const float* __restrict__ x, const float* __restrict__ g,
        const float* __restrict__ bb, const float* __restrict__ topw,
        const float* __restrict__ topb,
        _Float16* __restrict__ x16, float* __restrict__ topp) {
    __shared__ float red[8];
    int t = blockIdx.x;
    int tid = threadIdx.x;
    const float4 xv = ((const float4*)(x + (size_t)t * H_))[tid];
    float s  = xv.x + xv.y + xv.z + xv.w;
    float ss = xv.x * xv.x + xv.y * xv.y + xv.z * xv.z + xv.w * xv.w;
    s  = block_sum8(s, red);
    ss = block_sum8(ss, red);
    float mu   = s * (1.0f / H_);
    float var  = ss * (1.0f / H_) - mu * mu;
    float rstd = rsqrtf(var + 1e-5f);
    float4 gv = ((const float4*)g)[tid];
    float4 bv = ((const float4*)bb)[tid];
    float n0 = (xv.x - mu) * rstd * gv.x + bv.x;
    float n1 = (xv.y - mu) * rstd * gv.y + bv.y;
    float n2 = (xv.z - mu) * rstd * gv.z + bv.z;
    float n3 = (xv.w - mu) * rstd * gv.w + bv.w;
    v4h o;
    o[0] = (_Float16)n0; o[1] = (_Float16)n1;
    o[2] = (_Float16)n2; o[3] = (_Float16)n3;
    ((v4h*)(x16 + (size_t)t * H_))[tid] = o;
    float4 w0 = ((const float4*)topw)[tid];
    float4 w1 = ((const float4*)(topw + H_))[tid];
    float d0 = n0 * w0.x + n1 * w0.y + n2 * w0.z + n3 * w0.w;
    float d1 = n0 * w1.x + n1 * w1.y + n2 * w1.z + n3 * w1.w;
    d0 = block_sum8(d0, red);
    d1 = block_sum8(d1, red);
    if (tid == 0) {
        float l0 = d0 + topb[0], l1 = d1 + topb[1];
        float m  = fmaxf(l0, l1);
        float e0 = __expf(l0 - m), e1 = __expf(l1 - m);
        float inv = 1.0f / (e0 + e1);
        topp[t * 2 + 0] = e0 * inv;
        topp[t * 2 + 1] = e1 * inv;
    }
}

// C[M,N] = A[M,K] * W[N,K]^T + bias[N]; per-wave 32x32 tile, block tile 64x128.
__global__ __launch_bounds__(256) void gemm_kernel(
        const _Float16* __restrict__ A, const _Float16* __restrict__ Bw,
        const float* __restrict__ bias,
        _Float16* __restrict__ out16, float* __restrict__ out32,
        int M, int N, int K) {
    int wave = threadIdx.x >> 5;
    int lane = threadIdx.x & 31;
    int m0 = blockIdx.y * 64 + (wave >> 2) * 32;
    int n0 = blockIdx.x * 128 + (wave & 3) * 32;
    if (m0 >= M || n0 >= N) return;
    v8f acc00 = {}, acc01 = {}, acc10 = {}, acc11 = {};
    for (int k = 0; k < K; k += 32) {
        v16h a0 = load_a(A, m0,      k, K);
        v16h a1 = load_a(A, m0 + 16, k, K);
        v16h b0 = load_b(Bw, n0,      k, K);
        v16h b1 = load_b(Bw, n0 + 16, k, K);
        acc00 = wmma32(a0, b0, acc00);
        acc01 = wmma32(a0, b1, acc01);
        acc10 = wmma32(a1, b0, acc10);
        acc11 = wmma32(a1, b1, acc11);
    }
    int col = lane & 15, hi = lane >> 4;
#pragma unroll
    for (int mi = 0; mi < 2; ++mi) {
#pragma unroll
        for (int ni = 0; ni < 2; ++ni) {
            v8f acc = (mi == 0) ? (ni == 0 ? acc00 : acc01)
                                : (ni == 0 ? acc10 : acc11);
            int cb = n0 + ni * 16 + col;
            float bval = bias ? bias[cb] : 0.0f;
#pragma unroll
            for (int v = 0; v < 8; ++v) {
                int row = m0 + mi * 16 + v + 8 * hi;
                float val = acc[v] + bval;
                if (out32) out32[(size_t)row * N + cb] = val;
                if (out16) out16[(size_t)row * N + cb] = (_Float16)val;
            }
        }
    }
}

// V slice of qkv -> Vt[bh][d][s]  (so P*V can use the row-major B loader)
__global__ __launch_bounds__(256) void transpose_v_kernel(
        const _Float16* __restrict__ qkv, _Float16* __restrict__ vt) {
    __shared__ _Float16 tile[32][33];
    int bh = blockIdx.x;                 // b*8 + h
    int s0 = blockIdx.y * 32;
    int d0 = blockIdx.z * 32;
    int b = bh >> 3, h = bh & 7;
    int tx = threadIdx.x & 31, ty = threadIdx.x >> 5;
    const _Float16* src = qkv + (size_t)(b * S_) * 3072 + 2048 + h * 128;
#pragma unroll
    for (int j = 0; j < 4; ++j)
        tile[ty + j * 8][tx] = src[(size_t)(s0 + ty + j * 8) * 3072 + d0 + tx];
    __syncthreads();
    _Float16* dst = vt + (size_t)bh * 128 * S_;
#pragma unroll
    for (int j = 0; j < 4; ++j)
        dst[(size_t)(d0 + ty + j * 8) * S_ + s0 + tx] = tile[tx][ty + j * 8];
}

// Flash attention: 1 wave = 16 query rows; key blocks of 32; WMMA throughout.
__global__ __launch_bounds__(128) void attn_kernel(
        const _Float16* __restrict__ qkv, const _Float16* __restrict__ vt,
        _Float16* __restrict__ attn16) {
    __shared__ __align__(16) _Float16 psh[4][16][32];
    int bh = blockIdx.x; int b = bh >> 3, h = bh & 7;
    int wave = threadIdx.x >> 5, lane = threadIdx.x & 31;
    int col = lane & 15, hi = lane >> 4;
    int q0 = blockIdx.y * 64 + wave * 16;
    const _Float16* qb  = qkv + (size_t)(b * S_) * 3072 + h * 128;
    const _Float16* kb_ = qkv + (size_t)(b * S_) * 3072 + 1024 + h * 128;
    const _Float16* vtb = vt + (size_t)bh * 128 * S_;
    v16h qf[4];
#pragma unroll
    for (int kk = 0; kk < 4; ++kk) qf[kk] = load_a(qb, q0, kk * 32, 3072);
    v8f o[8] = {};
    float mrow[8], lrow[8];
#pragma unroll
    for (int v = 0; v < 8; ++v) { mrow[v] = -1e30f; lrow[v] = 0.0f; }
    const float scale = 0.088388347648318447f;   // 1/sqrt(128)

    for (int kblk = 0; kblk < S_; kblk += 32) {
        v8f s0 = {}, s1 = {};
#pragma unroll
        for (int kk = 0; kk < 4; ++kk)
            s0 = wmma32(qf[kk], load_b(kb_, kblk,      kk * 32, 3072), s0);
#pragma unroll
        for (int kk = 0; kk < 4; ++kk)
            s1 = wmma32(qf[kk], load_b(kb_, kblk + 16, kk * 32, 3072), s1);
#pragma unroll
        for (int v = 0; v < 8; ++v) {
            float a = s0[v] * scale, c = s1[v] * scale;
            float mx = fmaxf(a, c);
#pragma unroll
            for (int off = 8; off; off >>= 1) mx = fmaxf(mx, __shfl_xor(mx, off, 32));
            float mn = fmaxf(mrow[v], mx);
            float p0 = __expf(a - mn), p1 = __expf(c - mn);
            float rs = p0 + p1;
#pragma unroll
            for (int off = 8; off; off >>= 1) rs += __shfl_xor(rs, off, 32);
            float alpha = __expf(mrow[v] - mn);
            lrow[v] = lrow[v] * alpha + rs;
            mrow[v] = mn;
#pragma unroll
            for (int d = 0; d < 8; ++d) o[d][v] *= alpha;
            psh[wave][v + 8 * hi][col]      = (_Float16)p0;
            psh[wave][v + 8 * hi][col + 16] = (_Float16)p1;
        }
        asm volatile("" ::: "memory");           // keep DS store->load order
        union { v16h v; v8h h2[2]; } up;         // P as A-fragment (16x32)
        up.h2[0] = *(const v8h*)&psh[wave][col][hi * 8];
        up.h2[1] = *(const v8h*)&psh[wave][col][16 + hi * 8];
        v16h pf = up.v;
#pragma unroll
        for (int d = 0; d < 8; ++d)
            o[d] = wmma32(pf, load_b(vtb, d * 16, kblk, S_), o[d]);
        asm volatile("" ::: "memory");
    }
#pragma unroll
    for (int d = 0; d < 8; ++d) {
#pragma unroll
        for (int v = 0; v < 8; ++v) {
            int row = q0 + v + 8 * hi;
            float val = o[d][v] / lrow[v];
            attn16[(size_t)(b * S_ + row) * H_ + h * 128 + d * 16 + col] =
                (_Float16)val;
        }
    }
}

// Per-token: expert softmax, top-2 (first-occurrence tie-break), aux sums.
__global__ __launch_bounds__(256) void finalize_kernel(
        const float* __restrict__ hlog, const float* __restrict__ topp,
        float* __restrict__ outIdx, float* __restrict__ outProb,
        float* __restrict__ ppe_sum, float* __restrict__ hb_sum) {
    int wave = threadIdx.x >> 5, lane = threadIdx.x & 31;
    int t = blockIdx.x * 8 + wave;
    float tp0 = topp[t * 2 + 0], tp1 = topp[t * 2 + 1];
    int e0 = lane, e1 = lane + 32;
    float l0 = hlog[(size_t)t * 64 + e0] * tp0;
    float l1 = hlog[(size_t)t * 64 + e1] * tp1;
    float m = fmaxf(l0, l1);
#pragma unroll
    for (int off = 16; off; off >>= 1) m = fmaxf(m, __shfl_xor(m, off, 32));
    float x0 = __expf(l0 - m), x1 = __expf(l1 - m);
    float s = x0 + x1;
#pragma unroll
    for (int off = 16; off; off >>= 1) s += __shfl_xor(s, off, 32);
    float inv = 1.0f / s;
    float p0 = x0 * inv, p1 = x1 * inv;
    atomicAdd(&ppe_sum[e0], p0);
    atomicAdd(&ppe_sum[e1], p1);
    if (lane == 0) { atomicAdd(&hb_sum[0], tp0); atomicAdd(&hb_sum[1], tp1); }
    float bv = (p0 >= p1) ? p0 : p1;
    int   bi = (p0 >= p1) ? e0 : e1;
#pragma unroll
    for (int off = 16; off; off >>= 1) {
        float ov = __shfl_xor(bv, off, 32);
        int   oi = __shfl_xor(bi, off, 32);
        if (ov > bv || (ov == bv && oi < bi)) { bv = ov; bi = oi; }
    }
    float q0 = (e0 == bi) ? -1.0f : p0;
    float q1 = (e1 == bi) ? -1.0f : p1;
    float cv = (q0 >= q1) ? q0 : q1;
    int   ci = (q0 >= q1) ? e0 : e1;
#pragma unroll
    for (int off = 16; off; off >>= 1) {
        float ov = __shfl_xor(cv, off, 32);
        int   oi = __shfl_xor(ci, off, 32);
        if (ov > cv || (ov == cv && oi < ci)) { cv = ov; ci = oi; }
    }
    if (lane == 0) {
        float denom = bv + cv;
        outIdx[t * 2 + 0] = (float)bi;
        outIdx[t * 2 + 1] = (float)ci;
        outProb[t * 2 + 0] = bv / denom;
        outProb[t * 2 + 1] = cv / denom;
    }
}

__global__ __launch_bounds__(64) void aux_kernel(
        const float* __restrict__ ppe_sum, const float* __restrict__ hb_sum,
        float* __restrict__ out_aux) {
    __shared__ float sred[64];
    int tid = threadIdx.x;
    float ppe = ppe_sum[tid] * (1.0f / NT_);
    sred[tid] = ppe * logf(ppe * 64.0f + 1e-9f);
    __syncthreads();
    for (int s2 = 32; s2; s2 >>= 1) {
        if (tid < s2) sred[tid] += sred[tid + s2];
        __syncthreads();
    }
    if (tid == 0) {
        float hb0 = hb_sum[0] * (1.0f / NT_), hb1 = hb_sum[1] * (1.0f / NT_);
        float hl = hb0 * logf(hb0 * 2.0f + 1e-9f) +
                   hb1 * logf(hb1 * 2.0f + 1e-9f);
        out_aux[0] = sred[0] + 0.1f * hl;
    }
}

// ---------------------------------------------------------------- launcher
extern "C" void kernel_launch(void* const* d_in, const int* in_sizes, int n_in,
                              void* d_out, int out_size, void* d_ws, size_t ws_size,
                              hipStream_t stream) {
    (void)in_sizes; (void)n_in; (void)out_size; (void)ws_size;
    const float* x      = (const float*)d_in[0];
    const float* ln_g   = (const float*)d_in[1];
    const float* ln_b   = (const float*)d_in[2];
    const float* in_w   = (const float*)d_in[3];
    const float* in_b   = (const float*)d_in[4];
    const float* out_w  = (const float*)d_in[5];
    const float* out_b  = (const float*)d_in[6];
    const float* top_w  = (const float*)d_in[7];
    const float* top_b  = (const float*)d_in[8];
    const float* hier_w = (const float*)d_in[9];
    const float* hier_b = (const float*)d_in[10];
    float* out = (float*)d_out;

    char* wsb = (char*)d_ws;
    size_t off = 0;
    auto alloc = [&](size_t bytes) -> char* {
        char* p = wsb + off;
        off += (bytes + 255) & ~(size_t)255;
        return p;
    };
    _Float16* x16     = (_Float16*)alloc((size_t)NT_ * H_ * 2);
    _Float16* qkv16   = (_Float16*)alloc((size_t)NT_ * 3 * H_ * 2);
    _Float16* vt16    = (_Float16*)alloc((size_t)NT_ * H_ * 2);
    _Float16* attn16  = (_Float16*)alloc((size_t)NT_ * H_ * 2);
    _Float16* y16     = (_Float16*)alloc((size_t)NT_ * H_ * 2);
    _Float16* inw16   = (_Float16*)alloc((size_t)3 * H_ * H_ * 2);
    _Float16* outw16  = (_Float16*)alloc((size_t)H_ * H_ * 2);
    _Float16* hierw16 = (_Float16*)alloc((size_t)NE_ * H_ * 2);
    float* topp = (float*)alloc((size_t)NT_ * 2 * 4);
    float* hlog = (float*)alloc((size_t)NT_ * NE_ * 4);
    float* ppe  = (float*)alloc(66 * 4);
    float* hb   = ppe + 64;

    hipMemsetAsync(ppe, 0, 66 * 4, stream);

    cvt_f32_to_f16<<<(3 * H_ * H_ + 255) / 256, 256, 0, stream>>>(in_w, inw16, 3 * H_ * H_);
    cvt_f32_to_f16<<<(H_ * H_ + 255) / 256, 256, 0, stream>>>(out_w, outw16, H_ * H_);
    cvt_f32_to_f16<<<(NE_ * H_ + 255) / 256, 256, 0, stream>>>(hier_w, hierw16, NE_ * H_);

    ln_top_kernel<<<NT_, 256, 0, stream>>>(x, ln_g, ln_b, top_w, top_b, x16, topp);

    // QKV: [4096,1024] x [3072,1024]^T
    gemm_kernel<<<dim3(3 * H_ / 128, NT_ / 64), 256, 0, stream>>>(
        x16, inw16, in_b, qkv16, nullptr, NT_, 3 * H_, H_);

    transpose_v_kernel<<<dim3(B_ * NHEADS_, S_ / 32, DHEAD_ / 32), 256, 0, stream>>>(qkv16, vt16);

    attn_kernel<<<dim3(B_ * NHEADS_, S_ / 64), 128, 0, stream>>>(qkv16, vt16, attn16);

    // out-proj: [4096,1024] x [1024,1024]^T
    gemm_kernel<<<dim3(H_ / 128, NT_ / 64), 256, 0, stream>>>(
        attn16, outw16, out_b, y16, nullptr, NT_, H_, H_);

    // hierarchy routers: [4096,1024] x [64,1024]^T -> f32 logits
    gemm_kernel<<<dim3(1, NT_ / 64), 256, 0, stream>>>(
        y16, hierw16, hier_b, nullptr, hlog, NT_, NE_, H_);

    finalize_kernel<<<NT_ / 8, 256, 0, stream>>>(
        hlog, topp, out, out + NT_ * 2, ppe, hb);

    aux_kernel<<<1, 64, 0, stream>>>(ppe, hb, out + NT_ * 4);
}
